// ScaledDotProductAttention_19963007991923
// MI455X (gfx1250) — matrix operand order repaired and checked by reading, not compiled, not run on hardware
//
#include <hip/hip_runtime.h>

typedef __attribute__((ext_vector_type(16))) _Float16 v16h;
typedef __attribute__((ext_vector_type(8)))  _Float16 v8h;
typedef __attribute__((ext_vector_type(2)))  _Float16 v2h;
typedef __attribute__((ext_vector_type(8)))  float    v8f;
typedef __attribute__((ext_vector_type(4)))  float    v4f;

#define SDIM 2048
#define DKC  64
#define DVC  64
#define BQ   128      // queries per workgroup
#define WQ   16       // queries per wave
#define BKV  64       // keys per block iteration (2 x 32-deep WMMA chunks)
#define NWAVE 8
#define NTHREADS 256

// LDS strides in f16 units: 144B rows -> 16B-aligned, bank-conflict-free
#define K_STRIDE 72   // 64 dk + 8 pad
#define V_STRIDE 72   // 64 keys + 8 pad
#define P_STRIDE 72   // 64 keys + 8 pad

static __device__ inline v16h cat8(v8h lo, v8h hi) {
  return __builtin_shufflevector(lo, hi, 0,1,2,3,4,5,6,7,8,9,10,11,12,13,14,15);
}

__global__ __launch_bounds__(NTHREADS)
void fa_fwd_causal(const float* __restrict__ q, const float* __restrict__ k,
                   const float* __restrict__ v, float* __restrict__ out) {
  __shared__ __align__(16) _Float16 Klds[BKV * K_STRIDE];        // 64 keys x 64 dk (row-major)
  __shared__ __align__(16) _Float16 Vtlds[DVC * V_STRIDE];       // 64 dv x 64 keys (transposed)
  __shared__ __align__(16) _Float16 Plds[NWAVE * WQ * P_STRIDE]; // per-wave 16 x 64 probs

  const int tid  = threadIdx.x;
  const int lane = tid & 31;
  const int wave = tid >> 5;
  const int hl   = lane >> 4;    // half-wave: 0 or 1
  const int l16  = lane & 15;

  const int bh   = blockIdx.y;                  // b*H + h  (64 total)
  const int q0   = blockIdx.x * BQ + wave * WQ; // this wave's first query row

  const float* qg = q   + (size_t)bh * SDIM * DKC;
  const float* kg = k   + (size_t)bh * SDIM * DKC;
  const float* vg = v   + (size_t)bh * SDIM * DVC;
  float*       og = out + (size_t)bh * SDIM * DVC;

  // ---- Q tile (16x64) -> two f16 A-fragments ----
  // Scale folds 1/sqrt(dk) AND log2(e): softmax runs in exp2 domain, so every
  // exponent is a raw v_exp_f32 with no preceding multiply.
  v16h qa[2];
  {
    const int k0 = 8 * hl;
    const float sc = 0.125f * 1.44269504088896340736f;
    #pragma unroll
    for (int c = 0; c < 2; ++c) {
      const float* row = qg + (size_t)(q0 + l16) * DKC + 32 * c + k0;
      v4f x0 = *(const v4f*)(row + 0);
      v4f x1 = *(const v4f*)(row + 4);
      v4f x2 = *(const v4f*)(row + 16);
      v4f x3 = *(const v4f*)(row + 20);
      #pragma unroll
      for (int e = 0; e < 4; ++e) {
        qa[c][e]      = (_Float16)(x0[e] * sc);
        qa[c][e + 4]  = (_Float16)(x1[e] * sc);
        qa[c][e + 8]  = (_Float16)(x2[e] * sc);
        qa[c][e + 12] = (_Float16)(x3[e] * sc);
      }
    }
  }

  // Ones-column B fragment: B[k][0] = 1, else 0  ->  WMMA(P, ones) accumulates
  // per-row sums of P into C-layout column 0 (lanes 0 and 16).
  v16h ones;
  {
    const _Float16 ov = (l16 == 0) ? (_Float16)1.0f : (_Float16)0.0f;
    #pragma unroll
    for (int e = 0; e < 16; ++e) ones[e] = ov;
  }

  // Flash accumulators. C-layout: VGPR r <-> row (r + 8*hl), col = l16.
  v8f o_acc[4] = {{}, {}, {}, {}};
  v8f l_acc = {};                       // running softmax denominators (col 0)
  float m_i[8];
  #pragma unroll
  for (int r = 0; r < 8; ++r) m_i[r] = -__builtin_inff();

  const int nblocks = blockIdx.x * (BQ / BKV) + (BQ / BKV); // causal bound for this WG

  for (int kb_i = 0; kb_i < nblocks; ++kb_i) {
    const int kb = kb_i * BKV;

    // ---- cooperative K/V load: fp32 global -> f16 LDS (V transposed) ----
    #pragma unroll
    for (int pp = 0; pp < 2; ++pp) {
      const int row = (tid >> 3) + 32 * pp;  // 0..63 (key within block)
      const int col = (tid & 7) * 8;         // 0,8,...,56
      const float* ksrc = kg + (size_t)(kb + row) * DKC + col;
      v4f a0 = *(const v4f*)(ksrc + 0);
      v4f a1 = *(const v4f*)(ksrc + 4);
      v8h kh;
      #pragma unroll
      for (int e = 0; e < 4; ++e) { kh[e] = (_Float16)a0[e]; kh[e + 4] = (_Float16)a1[e]; }
      *(v8h*)&Klds[row * K_STRIDE + col] = kh;

      const float* vsrc = vg + (size_t)(kb + row) * DVC + col;
      v4f b0 = *(const v4f*)(vsrc + 0);
      v4f b1 = *(const v4f*)(vsrc + 4);
      #pragma unroll
      for (int e = 0; e < 4; ++e) {
        Vtlds[(col + e)     * V_STRIDE + row] = (_Float16)b0[e];
        Vtlds[(col + 4 + e) * V_STRIDE + row] = (_Float16)b1[e];
      }
      if (kb + BKV < SDIM) {            // gfx1250 global_prefetch for next block
        __builtin_prefetch(ksrc + BKV * DKC, 0, 3);
        __builtin_prefetch(vsrc + BKV * DVC, 0, 3);
      }
    }
    __syncthreads();

    const bool active = (kb <= q0 + WQ - 1);   // wave-uniform: EXEC stays all-ones
    if (active) {
      // ---- S = Q * K^T : four 16x16 f32 tiles over the 64-key block ----
      v8f s[4];
      #pragma unroll
      for (int f = 0; f < 4; ++f) {
        v8f acc = {};
        #pragma unroll
        for (int c = 0; c < 2; ++c) {
          // B frag: lane col n=l16 -> key row (16f+n); elems e<8 -> dk = 32c + 8*hl + e, e>=8 -> dk = 32c + 16 + 8*hl + (e-8)
          const _Float16* src = &Klds[(f * 16 + l16) * K_STRIDE + 32 * c + 8 * hl];
          v16h bfrag = cat8(*(const v8h*)src, *(const v8h*)(src + 16));
          acc = __builtin_amdgcn_wmma_f32_16x16x32_f16(false, qa[c], false, bfrag,
                                                       (short)0, acc, false, false);
        }
        s[f] = acc;
      }

      // ---- causal mask (only on diagonal blocks) ----
      if (kb + BKV - 1 > q0) {
        #pragma unroll
        for (int f = 0; f < 4; ++f) {
          const int kpos = kb + f * 16 + l16;
          #pragma unroll
          for (int r = 0; r < 8; ++r) {
            const int qpos = q0 + r + 8 * hl;
            s[f][r] = (kpos > qpos) ? -__builtin_inff() : s[f][r];
          }
        }
      }

      // ---- row max via packed-f16 half-wave reduction (2 rows per shfl) ----
      v2h mp[4];
      #pragma unroll
      for (int j = 0; j < 4; ++j) {
        float a0 = fmaxf(fmaxf(s[0][2 * j],     s[1][2 * j]),
                         fmaxf(s[2][2 * j],     s[3][2 * j]));
        float a1 = fmaxf(fmaxf(s[0][2 * j + 1], s[1][2 * j + 1]),
                         fmaxf(s[2][2 * j + 1], s[3][2 * j + 1]));
        mp[j][0] = (_Float16)a0;
        mp[j][1] = (_Float16)a1;
      }
      #pragma unroll
      for (int off = 1; off < 16; off <<= 1) {
        #pragma unroll
        for (int j = 0; j < 4; ++j) {
          int x = __shfl_xor(__builtin_bit_cast(int, mp[j]), off, 32);
          mp[j] = __builtin_elementwise_max(mp[j], __builtin_bit_cast(v2h, x));
        }
      }
      float mnew[8], alpha[8];
      #pragma unroll
      for (int r = 0; r < 8; ++r) {
        mnew[r]  = fmaxf(m_i[r], (float)mp[r >> 1][r & 1]);
        alpha[r] = __builtin_amdgcn_exp2f(m_i[r] - mnew[r]);
        m_i[r]   = mnew[r];
      }

      // ---- P = exp2(S - m) ----
      v8f p[4];
      #pragma unroll
      for (int f = 0; f < 4; ++f)
        #pragma unroll
        for (int r = 0; r < 8; ++r)
          p[f][r] = __builtin_amdgcn_exp2f(s[f][r] - mnew[r]);

      // ---- rescale running accumulators (O and l share alpha) ----
      #pragma unroll
      for (int t = 0; t < 4; ++t)
        #pragma unroll
        for (int r = 0; r < 8; ++r)
          o_acc[t][r] *= alpha[r];
      #pragma unroll
      for (int r = 0; r < 8; ++r) l_acc[r] *= alpha[r];

      // ---- P: C-layout -> f16 -> per-wave LDS -> A-fragments (2 chunks) ----
      _Float16* pw = &Plds[wave * (WQ * P_STRIDE)];
      #pragma unroll
      for (int f = 0; f < 4; ++f)
        #pragma unroll
        for (int r = 0; r < 8; ++r)
          pw[(r + 8 * hl) * P_STRIDE + f * 16 + l16] = (_Float16)p[f][r];
      v16h pa[2];
      #pragma unroll
      for (int c = 0; c < 2; ++c) {
        const _Float16* src = &pw[l16 * P_STRIDE + 32 * c + 8 * hl];
        pa[c] = cat8(*(const v8h*)src, *(const v8h*)(src + 16));
      }

      // ---- O += P * V (4 dv tiles x 2 chunks) and l += P * ones (2 WMMAs) ----
      #pragma unroll
      for (int t = 0; t < 4; ++t) {
        #pragma unroll
        for (int c = 0; c < 2; ++c) {
          const _Float16* vsrc = &Vtlds[(t * 16 + l16) * V_STRIDE + 32 * c + 8 * hl];
          v16h vfrag = cat8(*(const v8h*)vsrc, *(const v8h*)(vsrc + 16));
          o_acc[t] = __builtin_amdgcn_wmma_f32_16x16x32_f16(false, pa[c], false, vfrag,
                                                            (short)0, o_acc[t], false, false);
        }
      }
      #pragma unroll
      for (int c = 0; c < 2; ++c)
        l_acc = __builtin_amdgcn_wmma_f32_16x16x32_f16(false, pa[c], false, ones,
                                                       (short)0, l_acc, false, false);
    }
    __syncthreads();
  }

  // ---- epilogue: broadcast l from column-0 lanes, normalize, store fp32 ----
  float inv_l[8];
  #pragma unroll
  for (int r = 0; r < 8; ++r) {
    float lr = __shfl(l_acc[r], 16 * hl, 32);
    inv_l[r] = 1.0f / lr;
  }
  #pragma unroll
  for (int t = 0; t < 4; ++t)
    #pragma unroll
    for (int r = 0; r < 8; ++r)
      og[(size_t)(q0 + r + 8 * hl) * DVC + t * 16 + l16] = o_acc[t][r] * inv_l[r];
}

extern "C" void kernel_launch(void* const* d_in, const int* in_sizes, int n_in,
                              void* d_out, int out_size, void* d_ws, size_t ws_size,
                              hipStream_t stream) {
  (void)in_sizes; (void)n_in; (void)d_ws; (void)ws_size; (void)out_size;
  const float* q = (const float*)d_in[0];
  const float* k = (const float*)d_in[1];
  const float* v = (const float*)d_in[2];
  // d_in[3] is the causal mask; causality is computed analytically in-kernel.
  float* out = (float*)d_out;
  dim3 grid(SDIM / BQ, 4 * 16);   // 16 query blocks x (B*H)=64
  fa_fwd_causal<<<grid, NTHREADS, 0, stream>>>(q, k, v, out);
}
